// RuleNetwork_72155450573298
// MI455X (gfx1250) — matrix operand order repaired and checked by reading, not compiled, not run on hardware
//
#include <hip/hip_runtime.h>
#include <hip/hip_bf16.h>

#define H 1024
#define R 16
#define MT 32            // batch rows per workgroup in the GEMM kernel
#define WAVES 8          // 256 threads = 8 wave32
#define NT_PER_H (H / 16)        // 64 n-tiles of 16 cols
#define KC_PER_H (H / 32)        // 32 k-chunks of 32
// swizzled bf16 weight matrix size in dwords: rules * ntiles * kchunks * 32 lanes * 8 dwords
#define WMAT_DWORDS ((size_t)R * NT_PER_H * KC_PER_H * 32 * 8)

typedef __attribute__((ext_vector_type(16))) __bf16 v16bf;
typedef __attribute__((ext_vector_type(8)))  float  v8f;
typedef __attribute__((ext_vector_type(8)))  unsigned int v8u;
typedef __attribute__((ext_vector_type(4)))  unsigned int v4u;

// LDS: xs (64KB bf16 x-tile) + hs (64KB bf16 hidden tile) + acc (128KB f32) + wcs (2KB)
#define SMEM_BYTES (MT * H * 2 + MT * H * 2 + MT * H * 4 + MT * R * 4)

__device__ __forceinline__ unsigned int pack_bf16(float lo, float hi) {
    __bf16 l = (__bf16)lo;
    __bf16 h = (__bf16)hi;
    unsigned short lu = __builtin_bit_cast(unsigned short, l);
    unsigned short hu = __builtin_bit_cast(unsigned short, h);
    return ((unsigned int)hu << 16) | (unsigned int)lu;
}

// A fragment (16x32 bf16), rows [row0..row0+15], from LDS (base -> [row0][k0], stride ldw).
// Layout: lanes 0-15 hold K {0..15}, lanes 16-31 hold K {16..31} (matches B fragments).
// Loaded as two 16B vectors so the backend emits ds_load_b128.
__device__ __forceinline__ v16bf load_a_frag(const __bf16* base, int ldw, int lane) {
    int row = lane & 15;
    int kb  = (lane >> 4) << 4;            // 0 or 16
    const v4u* p0 = (const v4u*)(base + row * ldw + kb);        // 16B aligned
    const v4u* p1 = (const v4u*)(base + row * ldw + kb + 8);    // 16B aligned
    v4u a = *p0;
    v4u b = *p1;
    v8u u;
    u[0] = a[0]; u[1] = a[1]; u[2] = a[2]; u[3] = a[3];
    u[4] = b[0]; u[5] = b[1]; u[6] = b[2]; u[7] = b[3];
    return __builtin_bit_cast(v16bf, u);
}

// Direct (fallback) B fragment from fp32 row-major W[K][ldn]: convert on the fly.
__device__ __forceinline__ v16bf load_b_frag_f32(const float* __restrict__ W,
                                                 int k0, int n0, int ldn, int lane) {
    int col = n0 + (lane & 15);
    int kb  = k0 + ((lane >> 4) << 4);
    v8u u;
#pragma unroll
    for (int j = 0; j < 8; ++j) {
        float f0 = W[(kb + 2 * j)     * ldn + col];
        float f1 = W[(kb + 2 * j + 1) * ldn + col];
        u[j] = pack_bf16(f0, f1);
    }
    __builtin_prefetch(&W[(kb + 32) * ldn + col], 0, 3);
    return __builtin_bit_cast(v16bf, u);
}

// Swizzled B fragment: weights pre-packed per (rule, ntile, kchunk, lane, 8 dwords),
// so each lane loads 32 contiguous bytes (2x global_load_b128) with zero VALU.
__device__ __forceinline__ v16bf load_b_frag_swz(const unsigned int* __restrict__ base,
                                                 int nt, int kc, int lane) {
    const unsigned int* p = base + ((((size_t)nt * KC_PER_H + kc) * 32) + lane) * 8;
    v8u u = *(const v8u*)p;
    __builtin_prefetch(p + 32 * 8, 0, 3);   // next k-chunk (1KB ahead), near scope
    return __builtin_bit_cast(v16bf, u);
}

// ---------------------------------------------------------------------------
// Kernel 0: convert + swizzle W1/W2 fp32 -> bf16 in WMMA B-fragment order.
// One thread produces one lane's 8 dwords of one fragment.
// ---------------------------------------------------------------------------
__global__ void swizzle_weights_kernel(const float* __restrict__ W1,
                                       const float* __restrict__ W2,
                                       unsigned int* __restrict__ w1s,
                                       unsigned int* __restrict__ w2s) {
    size_t t = (size_t)blockIdx.x * blockDim.x + threadIdx.x;
    const size_t total = (size_t)2 * R * NT_PER_H * KC_PER_H * 32;
    if (t >= total) return;
    int lane = (int)(t & 31);
    size_t rest = t >> 5;
    int kc = (int)(rest & (KC_PER_H - 1)); rest >>= 5;
    int nt = (int)(rest & (NT_PER_H - 1)); rest >>= 6;
    int r  = (int)(rest & (R - 1));        rest >>= 4;
    int mat = (int)rest;                   // 0 = W1, 1 = W2

    const float* W = (mat ? W2 : W1) + (size_t)r * H * H;
    unsigned int* dst = (mat ? w2s : w1s) +
        (((((size_t)r * NT_PER_H + nt) * KC_PER_H + kc) * 32) + lane) * 8;

    int col = nt * 16 + (lane & 15);
    int kb  = kc * 32 + ((lane >> 4) << 4);
#pragma unroll
    for (int j = 0; j < 8; ++j) {
        dst[j] = pack_bf16(W[(size_t)(kb + 2 * j) * H + col],
                           W[(size_t)(kb + 2 * j + 1) * H + col]);
    }
}

// ---------------------------------------------------------------------------
// Kernel 1: gate (softmax) * confidence (sigmoid), renormalized weights.
// One wave32 per batch row.
// ---------------------------------------------------------------------------
__global__ void rule_gate_kernel(const float* __restrict__ x,
                                 const float* __restrict__ Wa,
                                 const float* __restrict__ ba,
                                 const float* __restrict__ Wc,
                                 const float* __restrict__ bc,
                                 float* __restrict__ wcomb,
                                 float* __restrict__ conf_out,
                                 int B) {
    int wave = (int)(threadIdx.x >> 5);
    int lane = (int)(threadIdx.x & 31);
    int b = (int)blockIdx.x * WAVES + wave;
    if (b >= B) return;

    float aA[R], aC[R];
#pragma unroll
    for (int r = 0; r < R; ++r) { aA[r] = 0.0f; aC[r] = 0.0f; }

    const float* xrow = x + (size_t)b * H;
    for (int h = lane; h < H; h += 32) {
        float xv = xrow[h];
#pragma unroll
        for (int r = 0; r < R; ++r) {
            aA[r] += xv * Wa[h * R + r];
            aC[r] += xv * Wc[r * H + h];
        }
    }
#pragma unroll
    for (int r = 0; r < R; ++r) {
#pragma unroll
        for (int off = 16; off > 0; off >>= 1) {
            aA[r] += __shfl_xor(aA[r], off, 32);
            aC[r] += __shfl_xor(aC[r], off, 32);
        }
    }
    float la[R], cf[R];
    float mx = -3.0e38f;
#pragma unroll
    for (int r = 0; r < R; ++r) {
        la[r] = aA[r] + ba[r];
        mx = fmaxf(mx, la[r]);
    }
    float se = 0.0f;
#pragma unroll
    for (int r = 0; r < R; ++r) { la[r] = __expf(la[r] - mx); se += la[r]; }
    float cs = 0.0f;
#pragma unroll
    for (int r = 0; r < R; ++r) {
        cf[r] = 1.0f / (1.0f + __expf(-(aC[r] + bc[r])));
        float cmb = (la[r] / se) * cf[r];
        la[r] = cmb;
        cs += cmb;
    }
    float inv = 1.0f / (cs + 1e-8f);
    if (lane == 0) {
#pragma unroll
        for (int r = 0; r < R; ++r) {
            wcomb[(size_t)b * R + r]    = la[r] * inv;
            conf_out[(size_t)b * R + r] = cf[r];
        }
    }
}

// ---------------------------------------------------------------------------
// Kernel 2: fused per-rule MLP (two chained GEMMs per rule), WMMA bf16.
// One workgroup per 32-row batch tile; loops over all 16 rules so the
// weighted output accumulator stays on-chip (LDS).
// SWZ=true  -> B fragments from pre-swizzled bf16 weights (fast path)
// SWZ=false -> B fragments converted from fp32 on the fly (fallback)
// ---------------------------------------------------------------------------
template <bool SWZ>
__global__ void rule_mlp_kernel(const float* __restrict__ x,
                                const float* __restrict__ W1,
                                const float* __restrict__ b1,
                                const float* __restrict__ W2,
                                const float* __restrict__ b2,
                                const unsigned int* __restrict__ w1s,
                                const unsigned int* __restrict__ w2s,
                                const float* __restrict__ wcomb,
                                float* __restrict__ out) {
    extern __shared__ char smem[];
    __bf16* xs  = (__bf16*)smem;                            // [MT][H] bf16
    __bf16* hs  = (__bf16*)(smem + MT * H * 2);             // [MT][H] bf16
    float*  acc = (float*)(smem + MT * H * 4);              // [MT][H] f32
    float*  wcs = (float*)(smem + MT * H * 4 + MT * H * 4); // [MT][R] f32

    const int m0   = (int)blockIdx.x * MT;
    const int tid  = (int)threadIdx.x;
    const int wave = tid >> 5;
    const int lane = tid & 31;

    for (int i = tid; i < MT * H; i += (int)blockDim.x) {
        int row = i >> 10;
        int col = i & (H - 1);
        xs[i]  = (__bf16)x[(size_t)(m0 + row) * H + col];
        acc[i] = 0.0f;
    }
    for (int i = tid; i < MT * R; i += (int)blockDim.x) {
        int row = i >> 4;
        wcs[i] = wcomb[(size_t)(m0 + row) * R + (i & (R - 1))];
    }
    __syncthreads();

    for (int r = 0; r < R; ++r) {
        const float* W1r = W1 + (size_t)r * H * H;
        const float* b1r = b1 + r * H;
        const float* W2r = W2 + (size_t)r * H * H;
        const float* b2r = b2 + r * H;
        const unsigned int* w1sr = w1s + (size_t)r * NT_PER_H * KC_PER_H * 32 * 8;
        const unsigned int* w2sr = w2s + (size_t)r * NT_PER_H * KC_PER_H * 32 * 8;

        // ---- phase 1: h = relu(x @ W1r + b1r), bf16 into LDS ----
        for (int t = 0; t < NT_PER_H / WAVES; ++t) {
            int nt = wave + t * WAVES;
            int n0 = nt * 16;
            v8f c0 = {};
            v8f c1 = {};
#pragma unroll 2
            for (int kc = 0; kc < KC_PER_H; ++kc) {
                int k0 = kc * 32;
                v16bf a0 = load_a_frag(xs + k0,          H, lane);
                v16bf a1 = load_a_frag(xs + 16 * H + k0, H, lane);
                v16bf bf = SWZ ? load_b_frag_swz(w1sr, nt, kc, lane)
                               : load_b_frag_f32(W1r, k0, n0, H, lane);
                c0 = __builtin_amdgcn_wmma_f32_16x16x32_bf16(false, a0, false, bf,
                                                             (short)0, c0, false, false);
                c1 = __builtin_amdgcn_wmma_f32_16x16x32_bf16(false, a1, false, bf,
                                                             (short)0, c1, false, false);
            }
            int n  = n0 + (lane & 15);
            int mr = (lane >> 4) << 3;
            float bias = b1r[n];
#pragma unroll
            for (int v = 0; v < 8; ++v) {
                float t0 = c0[v] + bias; t0 = t0 > 0.0f ? t0 : 0.0f;
                hs[(mr + v) * H + n] = (__bf16)t0;
                float t1 = c1[v] + bias; t1 = t1 > 0.0f ? t1 : 0.0f;
                hs[(16 + mr + v) * H + n] = (__bf16)t1;
            }
        }
        __syncthreads();   // hs fully written before phase 2 reads all K columns

        // ---- phase 2: acc += w[b,r] * (h @ W2r + b2r) ----
        for (int t = 0; t < NT_PER_H / WAVES; ++t) {
            int nt = wave + t * WAVES;
            int n0 = nt * 16;
            v8f c0 = {};
            v8f c1 = {};
#pragma unroll 2
            for (int kc = 0; kc < KC_PER_H; ++kc) {
                int k0 = kc * 32;
                v16bf a0 = load_a_frag(hs + k0,          H, lane);
                v16bf a1 = load_a_frag(hs + 16 * H + k0, H, lane);
                v16bf bf = SWZ ? load_b_frag_swz(w2sr, nt, kc, lane)
                               : load_b_frag_f32(W2r, k0, n0, H, lane);
                c0 = __builtin_amdgcn_wmma_f32_16x16x32_bf16(false, a0, false, bf,
                                                             (short)0, c0, false, false);
                c1 = __builtin_amdgcn_wmma_f32_16x16x32_bf16(false, a1, false, bf,
                                                             (short)0, c1, false, false);
            }
            int n  = n0 + (lane & 15);
            int mr = (lane >> 4) << 3;
            float bias = b2r[n];
#pragma unroll
            for (int v = 0; v < 8; ++v) {
                acc[(mr + v) * H + n]      += (c0[v] + bias) * wcs[(mr + v) * R + r];
                acc[(16 + mr + v) * H + n] += (c1[v] + bias) * wcs[(16 + mr + v) * R + r];
            }
        }
        __syncthreads();   // protect hs before next rule overwrites it
    }

    for (int i = tid; i < MT * H; i += (int)blockDim.x) {
        int row = i >> 10;
        int col = i & (H - 1);
        out[(size_t)(m0 + row) * H + col] = acc[i];
    }
}

extern "C" void kernel_launch(void* const* d_in, const int* in_sizes, int n_in,
                              void* d_out, int out_size, void* d_ws, size_t ws_size,
                              hipStream_t stream) {
    const float* x  = (const float*)d_in[0];
    const float* Wa = (const float*)d_in[1];
    const float* ba = (const float*)d_in[2];
    const float* W1 = (const float*)d_in[3];
    const float* b1 = (const float*)d_in[4];
    const float* W2 = (const float*)d_in[5];
    const float* b2 = (const float*)d_in[6];
    const float* Wc = (const float*)d_in[7];
    const float* bc = (const float*)d_in[8];

    const int B = in_sizes[0] / H;                 // 8192
    float* out  = (float*)d_out;                   // [B, H]
    float* conf = out + (size_t)B * H;             // [B, R]

    // workspace layout: wcomb [B,R] f32 | w1s bf16-swizzled | w2s bf16-swizzled
    float* wcomb = (float*)d_ws;
    size_t wcomb_bytes = (size_t)B * R * sizeof(float);
    unsigned int* w1s = (unsigned int*)((char*)d_ws + wcomb_bytes);
    unsigned int* w2s = w1s + WMAT_DWORDS;
    const size_t ws_needed = wcomb_bytes + 2 * WMAT_DWORDS * sizeof(unsigned int);
    const bool use_swz = (ws_size >= ws_needed);

    // Gate / confidence / renormalized weights
    rule_gate_kernel<<<(B + WAVES - 1) / WAVES, 32 * WAVES, 0, stream>>>(
        x, Wa, ba, Wc, bc, wcomb, conf, B);

    if (use_swz) {
        const size_t swz_threads = (size_t)2 * R * NT_PER_H * KC_PER_H * 32;
        swizzle_weights_kernel<<<(unsigned)((swz_threads + 255) / 256), 256, 0, stream>>>(
            W1, W2, w1s, w2s);
        (void)hipFuncSetAttribute((const void*)rule_mlp_kernel<true>,
                                  hipFuncAttributeMaxDynamicSharedMemorySize,
                                  SMEM_BYTES);
        rule_mlp_kernel<true><<<B / MT, 32 * WAVES, SMEM_BYTES, stream>>>(
            x, W1, b1, W2, b2, w1s, w2s, wcomb, out);
    } else {
        (void)hipFuncSetAttribute((const void*)rule_mlp_kernel<false>,
                                  hipFuncAttributeMaxDynamicSharedMemorySize,
                                  SMEM_BYTES);
        rule_mlp_kernel<false><<<B / MT, 32 * WAVES, SMEM_BYTES, stream>>>(
            x, W1, b1, W2, b2, w1s, w2s, wcomb, out);
    }
}
